// MHAAttention_40673340293162
// MI455X (gfx1250) — compile-verified, matrix operand order checked by reading, not run on hardware
//
#include <hip/hip_runtime.h>
#include <hip/hip_bf16.h>
#include <stdint.h>

// ---------------------------------------------------------------------------
// MHA with relative position bias (SAM/ViTDet-style block) for MI455X gfx1250.
// b=8, CH=512, H=W=32 (N=1024 tokens), HEADS=8, HEAD_DIM=64.
// All matmuls via v_wmma_f32_16x16x32_f16 (wave32).
//  - GEMMs: 32x64 per-wave register blocking; shared 64x32 weight tile staged
//    in LDS (double-buffered) via GLOBAL_LOAD_ASYNC_TO_LDS_B128 when the
//    builtin is available (ASYNCcnt path), ds-store fallback otherwise.
//  - Attention: flash-style, transposed-score trick (C-layout of S^T packs
//    directly into the f16 A-fragment of P), no LDS transpose needed.
// ---------------------------------------------------------------------------

typedef __attribute__((ext_vector_type(16))) _Float16 v16h;
typedef __attribute__((ext_vector_type(8)))  _Float16 v8h;
typedef __attribute__((ext_vector_type(8)))  float    v8f;
typedef __attribute__((ext_vector_type(4)))  int      v4i;

#define CH     512
#define NTOK   1024
#define HEADS  8
#define HD     64

#if defined(__has_builtin)
#if __has_builtin(__builtin_amdgcn_global_load_async_to_lds_b128) && \
    __has_builtin(__builtin_amdgcn_s_wait_asynccnt)
#define USE_ASYNC_LDS 1
#endif
#endif

#ifdef USE_ASYNC_LDS
// Per-lane async 16B copy global->LDS (tracked by ASYNCcnt).
// Builtin parameter type (from compiler diagnostic): v4i* (generic AS).
__device__ __forceinline__ void async_copy16(const _Float16* g, _Float16* l) {
  __builtin_amdgcn_global_load_async_to_lds_b128(
      (v4i*)(uintptr_t)g, (v4i*)(uintptr_t)l, 0, 0);
}
#endif

__device__ __forceinline__ v8f wmma_f16(v16h a, v16h b, v8f c) {
  return __builtin_amdgcn_wmma_f32_16x16x32_f16(
      /*neg_a=*/false, a, /*neg_b=*/false, b,
      /*c_mod=*/(short)0, c, /*reuse_a=*/false, /*reuse_b=*/false);
}

// Load a 16x32 (MxK) f16 A/B fragment. Lane L holds row/col M=L%16;
// lanes 0-15 carry K = {k0..k0+7, k0+16..k0+23}, lanes 16-31 carry
// K = {k0+8..k0+15, k0+24..k0+31} (2 halves per VGPR), per ISA 7.12.2.
// Works for both global and LDS base pointers (ds_load_b128 for shared).
__device__ __forceinline__ v16h load_frag16(const _Float16* __restrict__ base,
                                            int stride, int row, int k0, int lane) {
  const int m    = lane & 15;
  const int koff = (lane >> 4) << 3;  // 0 or 8
  const _Float16* p = base + (size_t)(row + m) * stride + k0 + koff;
  v8h lo = *(const v8h*)(p);
  v8h hi = *(const v8h*)(p + 16);
  v16h r;
#pragma unroll
  for (int t = 0; t < 8; ++t) { r[t] = lo[t]; r[t + 8] = hi[t]; }
  return r;
}

// ---------------- weight f32 -> f16 conversion --------------------------------
__global__ void k_f32_to_f16(const float* __restrict__ src, _Float16* __restrict__ dst, int n) {
  int i = blockIdx.x * blockDim.x + threadIdx.x;
  if (i < n) dst[i] = (_Float16)src[i];
}

// ---------------- LayerNorm over channels, emit f16 tokens --------------------
__global__ __launch_bounds__(256) void k_layernorm(const float* __restrict__ x,
                                                   const float* __restrict__ gw,
                                                   const float* __restrict__ gb,
                                                   _Float16* __restrict__ xn) {
  const int g = blockIdx.x * 256 + threadIdx.x;  // global token id (8192)
  const int b = g >> 10, i = g & 1023;
  const float* xb = x + (size_t)b * CH * NTOK + i;
  float s = 0.f, s2 = 0.f;
  for (int c = 0; c < CH; ++c) {
    float v = xb[(size_t)c * NTOK];
    s += v; s2 += v * v;
  }
  const float mu   = s * (1.f / CH);
  const float var  = s2 * (1.f / CH) - mu * mu;
  const float rstd = rsqrtf(var + 1e-6f);
  _Float16* o = xn + (size_t)g * CH;
  for (int c = 0; c < CH; ++c) {
    float v = (xb[(size_t)c * NTOK] - mu) * rstd * gw[c] + gb[c];
    o[c] = (_Float16)v;
  }
}

// ---------------- Q/K/V projection GEMM (M=8192, N=512, K=512) ----------------
// Per wave: 32x64 output (8 WMMA accs). Shared 64x32 weight tile staged in
// LDS per K-step (double-buffered, async when available). blockIdx.z = q/k/v.
// Q,K stored (b,h,n,d); V stored transposed (b,h,d,n). Scale folded into Q.
__global__ __launch_bounds__(256) void k_qkv(
    const _Float16* __restrict__ xn,
    const _Float16* __restrict__ wq16, const _Float16* __restrict__ wk16,
    const _Float16* __restrict__ wv16,
    const float* __restrict__ bq, const float* __restrict__ bk,
    const float* __restrict__ bv,
    _Float16* __restrict__ q, _Float16* __restrict__ kk,
    _Float16* __restrict__ vt) {
  __shared__ _Float16 bs[2][64 * 32];  // double-buffered B tile (2 x 4 KiB)

  const int z = blockIdx.z;
  const _Float16* w  = (z == 0) ? wq16 : (z == 1) ? wk16 : wv16;
  const float* bias  = (z == 0) ? bq   : (z == 1) ? bk   : bv;
  const int wave = threadIdx.x >> 5, lane = threadIdx.x & 31;
  const int hi = lane >> 4, ln = lane & 15;
  const int m0 = blockIdx.x * 256 + wave * 32;
  const int n0 = blockIdx.y * 64;

  // cooperative staging: thread t copies 16B (row t/4, 8-half chunk t%4)
  const int lrow = threadIdx.x >> 2;
  const int lchk = (threadIdx.x & 3) * 8;
  const _Float16* wsrc = w + (size_t)(n0 + lrow) * CH + lchk;
  _Float16* ldst = &bs[0][lrow * 32 + lchk];

  auto stage = [&](int k0, int buf) {
    const _Float16* src = wsrc + k0;
    _Float16* dst = ldst + buf * (64 * 32);
#ifdef USE_ASYNC_LDS
    async_copy16(src, dst);
#else
    *(v8h*)dst = *(const v8h*)src;
#endif
  };

  v8f zero = {};
  v8f acc[8];
#pragma unroll
  for (int t = 0; t < 8; ++t) acc[t] = zero;

  stage(0, 0);
  for (int kidx = 0; kidx < CH / 32; ++kidx) {
    const int k0 = kidx * 32;
#ifdef USE_ASYNC_LDS
    __builtin_amdgcn_s_wait_asynccnt(0);
#endif
    __syncthreads();
    if (kidx + 1 < CH / 32) stage(k0 + 32, (kidx + 1) & 1);

    if (k0 + 32 < CH)
      __builtin_prefetch(xn + (size_t)(m0 + ln) * CH + k0 + 32, 0, 1);
    const v16h a0 = load_frag16(xn, CH, m0,      k0, lane);
    const v16h a1 = load_frag16(xn, CH, m0 + 16, k0, lane);
    const _Float16* bb = &bs[kidx & 1][0];
#pragma unroll
    for (int t = 0; t < 4; ++t) {
      const v16h bf = load_frag16(bb, 32, t * 16, 0, lane);  // ds_load_b128
      acc[t]     = wmma_f16(a0, bf, acc[t]);
      acc[t + 4] = wmma_f16(a1, bf, acc[t + 4]);
    }
  }

  const float scale = (z == 0) ? 0.125f : 1.0f;  // 1/sqrt(HD) folded into Q
#pragma unroll
  for (int mt = 0; mt < 2; ++mt) {
#pragma unroll
    for (int t = 0; t < 4; ++t) {
      const v8f a = acc[mt * 4 + t];
      const int c = n0 + t * 16 + ln;
      const float bbv = bias[c];
      const int hh = c >> 6, dd = c & 63;
#pragma unroll
      for (int r = 0; r < 8; ++r) {
        const int g = m0 + mt * 16 + r + 8 * hi;
        const int b = g >> 10, i = g & 1023;
        const float v = (a[r] + bbv) * scale;
        if (z == 2)
          vt[(((size_t)(b * HEADS + hh)) * HD + dd) * NTOK + i] = (_Float16)v;
        else {
          _Float16* dst = (z == 0) ? q : kk;
          dst[(((size_t)(b * HEADS + hh)) * NTOK + i) * HD + dd] = (_Float16)v;
        }
      }
    }
  }
}

// ---------------- Flash attention with relative position bias -----------------
// grid = (b*heads, 1024/128). 8 waves/block, 16 query rows per wave.
// S^T = K @ Q^T so each lane tracks softmax stats for ONE query row (lane%16),
// and the f32 C-layout of S^T packs directly into the f16 A-fragment of P.
__global__ __launch_bounds__(256) void k_attn(
    const _Float16* __restrict__ Q, const _Float16* __restrict__ K,
    const _Float16* __restrict__ Vt, const float* __restrict__ rel,
    _Float16* __restrict__ O) {
  const int bh = blockIdx.x;
  const int h = bh & 7, b = bh >> 3;
  const int wave = threadIdx.x >> 5, lane = threadIdx.x & 31;
  const int hi = lane >> 4, ln = lane & 15;
  const int i0 = blockIdx.y * 128 + wave * 16;
  const int rowi = i0 + ln;                 // query row owned by this lane
  const int ri = rowi >> 5, rj = rowi & 31; // (y,x) coords of query token

  const _Float16* Qb = Q  + (size_t)bh * NTOK * HD;
  const _Float16* Kb = K  + (size_t)bh * NTOK * HD;
  const _Float16* Vb = Vt + (size_t)bh * HD * NTOK;

  const v16h bq0 = load_frag16(Qb, HD, i0, 0,  lane);  // Q^T as B-fragments
  const v16h bq1 = load_frag16(Qb, HD, i0, 32, lane);

  v8f o0 = {}, o1 = {}, o2 = {}, o3 = {};
  float rowmax = -1e30f, rowsum = 0.f;

  for (int j0 = 0; j0 < NTOK; j0 += 32) {
    // S^T tiles for key columns [j0,j0+16) and [j0+16,j0+32)
    v8f st0 = {}, st1 = {};
    st0 = wmma_f16(load_frag16(Kb, HD, j0, 0,  lane), bq0, st0);
    st0 = wmma_f16(load_frag16(Kb, HD, j0, 32, lane), bq1, st0);
    st1 = wmma_f16(load_frag16(Kb, HD, j0 + 16, 0,  lane), bq0, st1);
    st1 = wmma_f16(load_frag16(Kb, HD, j0 + 16, 32, lane), bq1, st1);

    // add decomposed relative position bias, track chunk max
    float s0[8], s1[8];
    float cmax = -1e30f;
#pragma unroll
    for (int r = 0; r < 8; ++r) {
      const int j  = j0 + r + 8 * hi;
      const int j2 = j + 16;
      const int idx0 = ((ri - (j  >> 5) + 31) * 63 + (rj - (j  & 31) + 31)) * HEADS + h;
      const int idx1 = ((ri - (j2 >> 5) + 31) * 63 + (rj - (j2 & 31) + 31)) * HEADS + h;
      s0[r] = st0[r] + rel[idx0];
      s1[r] = st1[r] + rel[idx1];
      cmax = fmaxf(cmax, fmaxf(s0[r], s1[r]));
    }
    cmax = fmaxf(cmax, __shfl_xor(cmax, 16, 32));  // combine the two col-halves
    const float newmax = fmaxf(rowmax, cmax);
    const float alpha  = __expf(rowmax - newmax);

    // P = exp(s - newmax), packed straight into the A-fragment layout
    float csum = 0.f;
    v16h pa;
#pragma unroll
    for (int r = 0; r < 8; ++r) {
      const float p0 = __expf(s0[r] - newmax);
      const float p1 = __expf(s1[r] - newmax);
      csum += p0 + p1;
      pa[r]     = (_Float16)p0;   // K =     r + 8*hi
      pa[r + 8] = (_Float16)p1;   // K = 16+ r + 8*hi
    }
    csum += __shfl_xor(csum, 16, 32);
    rowsum = rowsum * alpha + csum;
    rowmax = newmax;

    // rescale O accumulator rows (row r+8*hi) by alpha of that row
#pragma unroll
    for (int r = 0; r < 8; ++r) {
      const float a = __shfl(alpha, r + 8 * hi, 32);
      o0[r] *= a; o1[r] *= a; o2[r] *= a; o3[r] *= a;
    }
    // O += P @ V  (V^T stored (d, n): column dd is contiguous over j)
    o0 = wmma_f16(pa, load_frag16(Vb, NTOK,  0, j0, lane), o0);
    o1 = wmma_f16(pa, load_frag16(Vb, NTOK, 16, j0, lane), o1);
    o2 = wmma_f16(pa, load_frag16(Vb, NTOK, 32, j0, lane), o2);
    o3 = wmma_f16(pa, load_frag16(Vb, NTOK, 48, j0, lane), o3);
  }

  // normalize and emit f16 (b, n, heads*hd) for the output projection
  _Float16* Ob = O + (size_t)b * NTOK * CH;
#pragma unroll
  for (int r = 0; r < 8; ++r) {
    const int row = r + 8 * hi;
    const float rs = 1.f / __shfl(rowsum, row, 32);
    const size_t base = (size_t)(i0 + row) * CH + h * HD + ln;
    Ob[base +  0] = (_Float16)(o0[r] * rs);
    Ob[base + 16] = (_Float16)(o1[r] * rs);
    Ob[base + 32] = (_Float16)(o2[r] * rs);
    Ob[base + 48] = (_Float16)(o3[r] * rs);
  }
}

// ---------------- Output projection + residual (fp32 out) ---------------------
// Same LDS-staged, 32x64-per-wave GEMM structure as k_qkv.
__global__ __launch_bounds__(256) void k_proj(
    const _Float16* __restrict__ ao, const _Float16* __restrict__ wp16,
    const float* __restrict__ bp, const float* __restrict__ x,
    float* __restrict__ out) {
  __shared__ _Float16 bs[2][64 * 32];

  const int wave = threadIdx.x >> 5, lane = threadIdx.x & 31;
  const int hi = lane >> 4, ln = lane & 15;
  const int m0 = blockIdx.x * 256 + wave * 32;
  const int n0 = blockIdx.y * 64;

  const int lrow = threadIdx.x >> 2;
  const int lchk = (threadIdx.x & 3) * 8;
  const _Float16* wsrc = wp16 + (size_t)(n0 + lrow) * CH + lchk;
  _Float16* ldst = &bs[0][lrow * 32 + lchk];

  auto stage = [&](int k0, int buf) {
    const _Float16* src = wsrc + k0;
    _Float16* dst = ldst + buf * (64 * 32);
#ifdef USE_ASYNC_LDS
    async_copy16(src, dst);
#else
    *(v8h*)dst = *(const v8h*)src;
#endif
  };

  v8f zero = {};
  v8f acc[8];
#pragma unroll
  for (int t = 0; t < 8; ++t) acc[t] = zero;

  stage(0, 0);
  for (int kidx = 0; kidx < CH / 32; ++kidx) {
    const int k0 = kidx * 32;
#ifdef USE_ASYNC_LDS
    __builtin_amdgcn_s_wait_asynccnt(0);
#endif
    __syncthreads();
    if (kidx + 1 < CH / 32) stage(k0 + 32, (kidx + 1) & 1);

    if (k0 + 32 < CH)
      __builtin_prefetch(ao + (size_t)(m0 + ln) * CH + k0 + 32, 0, 1);
    const v16h a0 = load_frag16(ao, CH, m0,      k0, lane);
    const v16h a1 = load_frag16(ao, CH, m0 + 16, k0, lane);
    const _Float16* bb = &bs[kidx & 1][0];
#pragma unroll
    for (int t = 0; t < 4; ++t) {
      const v16h bf = load_frag16(bb, 32, t * 16, 0, lane);
      acc[t]     = wmma_f16(a0, bf, acc[t]);
      acc[t + 4] = wmma_f16(a1, bf, acc[t + 4]);
    }
  }

#pragma unroll
  for (int mt = 0; mt < 2; ++mt) {
#pragma unroll
    for (int t = 0; t < 4; ++t) {
      const v8f a = acc[mt * 4 + t];
      const int c = n0 + t * 16 + ln;
      const float bbv = bp[c];
#pragma unroll
      for (int r = 0; r < 8; ++r) {
        const int g = m0 + mt * 16 + r + 8 * hi;
        const int b = g >> 10, i = g & 1023;
        const size_t off = ((size_t)b * CH + c) * NTOK + i;
        out[off] = x[off] + a[r] + bbv;   // residual add
      }
    }
  }
}

// ---------------------------------------------------------------------------
extern "C" void kernel_launch(void* const* d_in, const int* in_sizes, int n_in,
                              void* d_out, int out_size, void* d_ws, size_t ws_size,
                              hipStream_t stream) {
  const float* x   = (const float*)d_in[0];
  const float* nw  = (const float*)d_in[1];
  const float* nb  = (const float*)d_in[2];
  const float* wq  = (const float*)d_in[3];
  const float* bq  = (const float*)d_in[4];
  const float* wk  = (const float*)d_in[5];
  const float* bk  = (const float*)d_in[6];
  const float* wv  = (const float*)d_in[7];
  const float* bv  = (const float*)d_in[8];
  const float* wp  = (const float*)d_in[9];
  const float* bp  = (const float*)d_in[10];
  const float* rel = (const float*)d_in[11];
  float* out = (float*)d_out;

  // workspace carve-out (42 MiB)
  char* ws = (char*)d_ws;
  _Float16* xn   = (_Float16*)(ws);                     // 8 MiB (8192 x 512)
  _Float16* q    = (_Float16*)(ws + (8u  << 20));       // 8 MiB (b,h,n,d)
  _Float16* kk   = (_Float16*)(ws + (16u << 20));       // 8 MiB (b,h,n,d)
  _Float16* vt   = (_Float16*)(ws + (24u << 20));       // 8 MiB (b,h,d,n)
  _Float16* ao   = (_Float16*)(ws + (32u << 20));       // 8 MiB (b,n,512)
  _Float16* wq16 = (_Float16*)(ws + (40u << 20));       // 4 x 512 KiB
  _Float16* wk16 = wq16 + CH * CH;
  _Float16* wv16 = wk16 + CH * CH;
  _Float16* wp16 = wv16 + CH * CH;

  const int WN = CH * CH;  // 262144
  k_f32_to_f16<<<WN / 256, 256, 0, stream>>>(wq, wq16, WN);
  k_f32_to_f16<<<WN / 256, 256, 0, stream>>>(wk, wk16, WN);
  k_f32_to_f16<<<WN / 256, 256, 0, stream>>>(wv, wv16, WN);
  k_f32_to_f16<<<WN / 256, 256, 0, stream>>>(wp, wp16, WN);

  k_layernorm<<<32, 256, 0, stream>>>(x, nw, nb, xn);   // 8192 tokens

  dim3 gq(32, 8, 3);   // M=8192/256, N=512/64, z = q/k/v
  k_qkv<<<gq, 256, 0, stream>>>(xn, wq16, wk16, wv16, bq, bk, bv, q, kk, vt);

  dim3 ga(64, 8);      // b*heads, 1024/128 query-row tiles
  k_attn<<<ga, 256, 0, stream>>>(q, kk, vt, rel, ao);

  dim3 gp(32, 8);      // M=8192/256, N=512/64
  k_proj<<<gp, 256, 0, stream>>>(ao, wp16, bp, x, out);
}
